// UniCoreLinearGroupFP4_8504035246564
// MI455X (gfx1250) — compile-verified, matrix operand order checked by reading, not run on hardware
//
#include <hip/hip_runtime.h>
#include <hip/hip_bf16.h>

typedef __attribute__((ext_vector_type(16))) __bf16 v16bf;
typedef __attribute__((ext_vector_type(8)))  __bf16 v8bf;
typedef __attribute__((ext_vector_type(8)))  float  v8f;

// Problem constants (from reference setup_inputs)
#define M_TOT 64      // B*S = 4*16
#define K_    4096
#define O_    11008
#define NG_   32      // K/G = 4096/128
#define KC    512     // K-chunk staged in LDS (multiple of G=128 and of 32)
#define PADE  8       // bf16 pad elements per LDS column row (16B) -> bank spread
#define LCOL  (KC + PADE)   // 520 elems per column row in LDS
#define NCHUNK (K_ / KC)    // 8
#define WAVES 4
#define TPB   (WAVES * 32)  // 128 threads

__global__ __launch_bounds__(TPB)
void dequant_gemm_bf16_wmma(const __bf16* __restrict__ x,
                            const __bf16* __restrict__ w,
                            const __bf16* __restrict__ wsc,
                            __bf16* __restrict__ out)
{
    // double-buffered dequantized weight tile: 2 bufs x 16 cols x LCOL bf16
    __shared__ __align__(16) __bf16 smem[2 * 16 * LCOL];

    const int tid   = threadIdx.x;
    const int wave  = tid >> 5;
    const int lane  = tid & 31;
    const int obase = blockIdx.x * 16;

    // -------- loader mapping: 16 cols x 8 sub-ranges of 64 k each --------
    const int  lcol = tid & 15;      // lanes 0..15 -> distinct cols (bank-friendly)
    const int  lsub = tid >> 4;      // 0..7
    const long wrow = (long)(obase + lcol) * K_;

    v8bf   stage[8];                 // 64 bf16 of weight per thread
    __bf16 sc;

    // prefetch chunk 0
    {
        const __bf16* p = w + wrow + lsub * 64;
        #pragma unroll
        for (int j = 0; j < 8; ++j)
            stage[j] = __builtin_nontemporal_load((const v8bf*)(p + j * 8));
        sc = wsc[(obase + lcol) * NG_ + ((lsub * 64) >> 7)];
    }

    // -------- per-wave WMMA operand addressing --------
    v8f acc = {};
    const int     mrow  = wave * 16 + (lane & 15);          // A row for this lane
    const int     khalf = (lane >> 4) ? 8 : 0;              // K split per ISA layout
    const __bf16* xrow  = x + (long)mrow * K_ + khalf;
    const int     bboff = (lane & 15) * LCOL + khalf;       // B column base in LDS

    int buf = 0;
    for (int c = 0; c < NCHUNK; ++c) {
        // dequantize staged registers -> LDS[buf] (packed bf16 multiply)
        {
            v8bf s8 = { sc, sc, sc, sc, sc, sc, sc, sc };
            __bf16* dst = smem + buf * 16 * LCOL + lcol * LCOL + lsub * 64;
            #pragma unroll
            for (int j = 0; j < 8; ++j) {
                v8bf dq = stage[j] * s8;
                *(v8bf*)(dst + j * 8) = dq;
            }
        }
        __syncthreads();

        // prefetch next chunk into registers (overlaps with WMMA below)
        if (c + 1 < NCHUNK) {
            const __bf16* p = w + wrow + (c + 1) * KC + lsub * 64;
            #pragma unroll
            for (int j = 0; j < 8; ++j)
                stage[j] = __builtin_nontemporal_load((const v8bf*)(p + j * 8));
            sc = wsc[(obase + lcol) * NG_ + (((c + 1) * KC + lsub * 64) >> 7)];
        }

        // compute: KC/32 = 16 WMMA steps over this chunk
        const __bf16* xc = xrow + c * KC;
        const __bf16* bb = smem + buf * 16 * LCOL + bboff;
        #pragma unroll
        for (int ks = 0; ks < KC / 32; ++ks) {
            const int kb = ks * 32;
            // A fragment: 16x32 bf16, lane<16 K{0..7,16..23}, lane>=16 K{8..15,24..31}
            v8bf alo = *(const v8bf*)(xc + kb);
            v8bf ahi = *(const v8bf*)(xc + kb + 16);
            v16bf a = __builtin_shufflevector(alo, ahi,
                        0,1,2,3,4,5,6,7,8,9,10,11,12,13,14,15);
            // B fragment: 32x16 bf16 from LDS (ds_load_b128 x2)
            v8bf blo = *(const v8bf*)(bb + kb);
            v8bf bhi = *(const v8bf*)(bb + kb + 16);
            v16bf bm = __builtin_shufflevector(blo, bhi,
                        0,1,2,3,4,5,6,7,8,9,10,11,12,13,14,15);
            acc = __builtin_amdgcn_wmma_f32_16x16x32_bf16(
                      false, a, false, bm, (short)0, acc, false, false);
        }
        __syncthreads();
        buf ^= 1;
    }

    // D layout: VGPR r -> M = wave*16 + r (+8 for lanes 16..31), N = lane&15
    const int mout = wave * 16 + ((lane >> 4) ? 8 : 0);
    const int ocol = obase + (lane & 15);
    #pragma unroll
    for (int r = 0; r < 8; ++r)
        out[(long)(mout + r) * O_ + ocol] = (__bf16)acc[r];
}

extern "C" void kernel_launch(void* const* d_in, const int* in_sizes, int n_in,
                              void* d_out, int out_size, void* d_ws, size_t ws_size,
                              hipStream_t stream) {
    const __bf16* x   = (const __bf16*)d_in[0];  // (4,16,4096) bf16
    const __bf16* w   = (const __bf16*)d_in[1];  // (11008,4096) bf16
    const __bf16* wsc = (const __bf16*)d_in[2];  // (11008,32) bf16
    __bf16*       out = (__bf16*)d_out;          // (4,16,11008) bf16

    dim3 grid(O_ / 16);   // 688 workgroups, one 16-col O tile each
    dim3 block(TPB);      // 4 waves: one M-tile per wave
    hipLaunchKernelGGL(dequant_gemm_bf16_wmma, grid, block, 0, stream,
                       x, w, wsc, out);
}
